// TransformerBlock_29463475651075
// MI455X (gfx1250) — matrix operand order repaired and checked by reading, not compiled, not run on hardware
//
#include <hip/hip_runtime.h>
#include <hip/hip_bf16.h>
#include <math.h>

// ---------------------------------------------------------------------------
// Transformer block for MI455X (gfx1250): bf16 WMMA GEMMs + flash attention,
// async global->LDS double-buffered staging where the toolchain supports it.
// Shapes (fixed per reference): B=4, S=2048, H=1024, NH=16, HD=64, E=4096.
// ---------------------------------------------------------------------------

typedef __attribute__((ext_vector_type(16))) __bf16 v16bf;
typedef __attribute__((ext_vector_type(8)))  float  v8f;

union Frag16 { v16bf v; unsigned int u[8]; __bf16 h[16]; };

__device__ __forceinline__ __bf16 f2bf(float x) {
  union { float f; unsigned int u; } a; a.f = x;
  unsigned int r = a.u + 0x7FFFu + ((a.u >> 16) & 1u);   // round-to-nearest-even
  union { unsigned short s; __bf16 h; } o; o.s = (unsigned short)(r >> 16);
  return o.h;
}

// A-matrix (16x32 bf16) dword index within a 16-dword (32 elem) K-row:
// lanes 0-15 hold K=0..15, lanes 16-31 K=16..31.
__device__ __forceinline__ int a_dw(int j, int half) { return j + half * 8; }
// B-matrix (32x16 bf16): lanes 0-15 hold K=0..15, lanes 16-31 K=16..31.
__device__ __forceinline__ int b_dw(int j, int half) { return j + half * 8; }

__device__ __forceinline__ v8f wmma_bf16(const Frag16& a, const Frag16& b, v8f c) {
  return __builtin_amdgcn_wmma_f32_16x16x32_bf16(false, a.v, false, b.v, (short)0, c,
                                                 false, false);
}

// ------------------------- async global -> LDS path -------------------------
#if __has_builtin(__builtin_amdgcn_global_load_async_to_lds_b128) && \
    __has_builtin(__builtin_amdgcn_s_wait_asynccnt)
#define HAVE_ASYNC_LDS 1
#else
#define HAVE_ASYNC_LDS 0
#endif

typedef int v4i_vs __attribute__((vector_size(16)));
typedef __attribute__((address_space(1))) v4i_vs* g_v4i_ptr;
typedef __attribute__((address_space(3))) v4i_vs* l_v4i_ptr;

__device__ __forceinline__ void ld_async_b128(void* lds, const void* g) {
#if HAVE_ASYNC_LDS
  __builtin_amdgcn_global_load_async_to_lds_b128((g_v4i_ptr)g, (l_v4i_ptr)lds, 0, 0);
#else
  *(uint4*)lds = *(const uint4*)g;
#endif
}
__device__ __forceinline__ void wait_async_all() {
#if HAVE_ASYNC_LDS
  __builtin_amdgcn_s_wait_asynccnt(0);
#endif
}

// ---------------------------------------------------------------------------
// Weight convert + transpose: dst[n*K + k] = bf16(src[k*N + n])
// ---------------------------------------------------------------------------
__global__ __launch_bounds__(256)
void cvtT_kernel(const float* __restrict__ src, __bf16* __restrict__ dst, int K, int N) {
  int n = blockIdx.x * 256 + threadIdx.x;
  int k = blockIdx.y;
  dst[(size_t)n * K + k] = f2bf(src[(size_t)k * N + n]);
}

// ---------------------------------------------------------------------------
// LayerNorm (H=1024) -> bf16, one row per 256-thread block
// ---------------------------------------------------------------------------
__global__ __launch_bounds__(256)
void ln_bf16_kernel(const float* __restrict__ x, const float* __restrict__ w,
                    const float* __restrict__ b, __bf16* __restrict__ out) {
  const int Hn = 1024;
  const int row = blockIdx.x;
  const int tid = threadIdx.x;
  const float* xr = x + (size_t)row * Hn;
  float4 v = ((const float4*)xr)[tid];
  float s  = v.x + v.y + v.z + v.w;
  float s2 = v.x * v.x + v.y * v.y + v.z * v.z + v.w * v.w;
#pragma unroll
  for (int off = 16; off > 0; off >>= 1) {
    s  += __shfl_xor(s,  off);
    s2 += __shfl_xor(s2, off);
  }
  __shared__ float ps[8], ps2[8];
  if ((tid & 31) == 0) { ps[tid >> 5] = s; ps2[tid >> 5] = s2; }
  __syncthreads();
  float ts = 0.f, ts2 = 0.f;
#pragma unroll
  for (int i = 0; i < 8; ++i) { ts += ps[i]; ts2 += ps2[i]; }
  const float mu   = ts * (1.0f / Hn);
  const float var  = ts2 * (1.0f / Hn) - mu * mu;
  const float rstd = rsqrtf(var + 1e-5f);
  float4 wv = ((const float4*)w)[tid];
  float4 bv = ((const float4*)b)[tid];
  float o0 = (v.x - mu) * rstd * wv.x + bv.x;
  float o1 = (v.y - mu) * rstd * wv.y + bv.y;
  float o2 = (v.z - mu) * rstd * wv.z + bv.z;
  float o3 = (v.w - mu) * rstd * wv.w + bv.w;
  union { __bf16 h[4]; uint2 u2; } pk;
  pk.h[0] = f2bf(o0); pk.h[1] = f2bf(o1); pk.h[2] = f2bf(o2); pk.h[3] = f2bf(o3);
  ((uint2*)(out + (size_t)row * Hn))[tid] = pk.u2;
}

// ---------------------------------------------------------------------------
// V transpose: Vt[((b*NH+h)*HD + d)*S + s] = qkv[(b*S+s)*3072 + 2048 + h*64 + d]
// ---------------------------------------------------------------------------
__global__ __launch_bounds__(256)
void vtrans_kernel(const __bf16* __restrict__ qkv, __bf16* __restrict__ Vt) {
  const int S = 2048;
  int s  = blockIdx.x * 256 + threadIdx.x;
  int d  = blockIdx.y;                    // 0..63
  int bh = blockIdx.z;                    // b*16 + h
  int b  = bh >> 4, h = bh & 15;
  Vt[((size_t)bh * 64 + d) * S + s] =
      qkv[((size_t)(b * S + s)) * 3072 + 2048 + h * 64 + d];
}

// ---------------------------------------------------------------------------
// Tiled bf16 WMMA GEMM: C[M x N] = A[M x K] @ Wt[N x K]^T + bias, epilogue.
// Block 256 thr (8 waves), tile 128x128, K-step 32, double-buffered LDS with
// async global->LDS staging. Each wave computes a 32x64 sub-tile (8 WMMAs).
// ---------------------------------------------------------------------------
enum { OP_BIAS_BF16 = 0, OP_GELU_BF16 = 1, OP_RESID_F32 = 2 };

template <int EPI>
__global__ __launch_bounds__(256)
void gemm_wmma_kernel(const __bf16* __restrict__ A, const __bf16* __restrict__ Wt,
                      const float* __restrict__ bias, const float* __restrict__ resid,
                      void* __restrict__ out, int M, int N, int K) {
  __shared__ unsigned int sA[2][128 * 16];   // 128 rows x 32 bf16, double buffered
  __shared__ unsigned int sB[2][128 * 16];   // 128 cols x 32 bf16
  const int tid  = threadIdx.x;
  const int lane = tid & 31;
  const int wave = tid >> 5;
  const int half = lane >> 4;
  const int l15  = lane & 15;
  const int wm   = wave & 3;                 // M strip (32 rows)
  const int wn   = wave >> 2;                // N half (64 cols)
  const int m0   = blockIdx.x * 128;
  const int n0   = blockIdx.y * 128;

  // staging coords: each thread moves 32B of A and 32B of B per K-step
  const int lrow = tid >> 1, lc = (tid & 1) * 8;   // lc in dwords (0 or 8)
  const __bf16* gA = A  + (size_t)(m0 + lrow) * K + lc * 2;
  const __bf16* gB = Wt + (size_t)(n0 + lrow) * K + lc * 2;

  auto stage = [&](int buf, int k0) {
    ld_async_b128(&sA[buf][lrow * 16 + lc],     gA + k0);
    ld_async_b128(&sA[buf][lrow * 16 + lc + 4], gA + k0 + 8);
    ld_async_b128(&sB[buf][lrow * 16 + lc],     gB + k0);
    ld_async_b128(&sB[buf][lrow * 16 + lc + 4], gB + k0 + 8);
  };

  v8f acc[2][4] = {};
  stage(0, 0);
  for (int k0 = 0; k0 < K; k0 += 32) {
    const int buf = (k0 >> 5) & 1;
    wait_async_all();            // our stage(buf) loads have landed in LDS
    __syncthreads();             // everyone's have
    if (k0 + 32 < K) stage(buf ^ 1, k0 + 32);   // overlap next stage w/ compute

    Frag16 a[2];
#pragma unroll
    for (int i = 0; i < 2; ++i)
#pragma unroll
      for (int j = 0; j < 8; ++j)
        a[i].u[j] = sA[buf][(wm * 32 + i * 16 + l15) * 16 + a_dw(j, half)];
#pragma unroll
    for (int t = 0; t < 4; ++t) {
      Frag16 b;
#pragma unroll
      for (int j = 0; j < 8; ++j)
        b.u[j] = sB[buf][(wn * 64 + t * 16 + l15) * 16 + b_dw(j, half)];
      acc[0][t] = wmma_bf16(a[0], b, acc[0][t]);
      acc[1][t] = wmma_bf16(a[1], b, acc[1][t]);
    }
    __syncthreads();             // all waves done with buf before it is reused
  }

  // D layout: lane holds col = t*16 + l15; rows = i*16 + half*8 + r
#pragma unroll
  for (int i = 0; i < 2; ++i) {
#pragma unroll
    for (int t = 0; t < 4; ++t) {
      const int col = n0 + wn * 64 + t * 16 + l15;
      const float bv = bias[col];
#pragma unroll
      for (int r = 0; r < 8; ++r) {
        const int m = m0 + wm * 32 + i * 16 + half * 8 + r;
        float v = acc[i][t][r] + bv;
        if (EPI == OP_BIAS_BF16) {
          ((__bf16*)out)[(size_t)m * N + col] = f2bf(v);
        } else if (EPI == OP_GELU_BF16) {
          v = 0.5f * v * (1.0f + erff(v * 0.70710678118654752f));
          ((__bf16*)out)[(size_t)m * N + col] = f2bf(v);
        } else {
          ((float*)out)[(size_t)m * N + col] = resid[(size_t)m * N + col] + v;
        }
      }
    }
  }
}

// ---------------------------------------------------------------------------
// Flash attention. Block = 128 thr (4 waves), covers 64 q rows; grid
// (S/64, NH, B). Computes S^T = K·Q^T per 32-key tile (lane owns one query),
// online softmax with lane-local (m,l), then O^T += V^T · P^T via WMMA.
// K/V tiles double-buffered in LDS with async staging.
// ---------------------------------------------------------------------------
__global__ __launch_bounds__(128)
void attn_kernel(const __bf16* __restrict__ qkv, const __bf16* __restrict__ Vt,
                 __bf16* __restrict__ o) {
  const int S = 2048, NH = 16, H3 = 3072, Hn = 1024;
  const int qt   = blockIdx.x;
  const int h    = blockIdx.y;
  const int bb   = blockIdx.z;
  const int tid  = threadIdx.x;
  const int lane = tid & 31;
  const int wave = tid >> 5;
  const int half = lane >> 4;
  const int l15  = lane & 15;

  __shared__ unsigned int sK[2][32 * 32];   // 32 keys x 64 d (bf16)
  __shared__ unsigned int sV[2][64 * 16];   // 64 d x 32 keys (transposed V)

  const int qrow = qt * 64 + wave * 16 + l15;       // this lane's query
  const __bf16* Qrow = qkv + ((size_t)(bb * S + qrow)) * H3 + h * 64;
  const unsigned int* Qd = (const unsigned int*)Qrow;
  Frag16 bq0, bq1;                                   // Q^T B-fragments (d 0-31, 32-63)
#pragma unroll
  for (int j = 0; j < 8; ++j) {
    bq0.u[j] = Qd[b_dw(j, half)];
    bq1.u[j] = Qd[16 + b_dw(j, half)];
  }

  const __bf16* Kbase = qkv + ((size_t)(bb * S)) * H3 + 1024 + h * 64;
  const __bf16* Vtb   = Vt + ((size_t)((bb * NH + h) * 64)) * S;

  auto stage = [&](int buf, int kb) {
#pragma unroll
    for (int it = tid; it < 256; it += 128) {
      int kc = it >> 3, c = (it & 7) * 4;
      ld_async_b128(&sK[buf][kc * 32 + c], Kbase + (size_t)(kb + kc) * H3 + c * 2);
    }
#pragma unroll
    for (int it = tid; it < 256; it += 128) {
      int d = it >> 2, c = (it & 3) * 4;
      ld_async_b128(&sV[buf][d * 16 + c], Vtb + (size_t)d * S + kb + c * 2);
    }
  };

  v8f oacc[4] = {};
  float mrow = -3.0e38f, lrow = 0.0f;

  stage(0, 0);
  for (int kb = 0; kb < S; kb += 32) {
    const int buf = (kb >> 5) & 1;
    wait_async_all();
    __syncthreads();
    if (kb + 32 < S) stage(buf ^ 1, kb + 32);

    // S^T = K · Q^T  (two 16-key D-fragments, each K-dim 64 = 2 WMMAs)
    Frag16 a;
    v8f st0 = {}, st1 = {};
#pragma unroll
    for (int j = 0; j < 8; ++j) a.u[j] = sK[buf][l15 * 32 + a_dw(j, half)];
    st0 = wmma_bf16(a, bq0, st0);
#pragma unroll
    for (int j = 0; j < 8; ++j) a.u[j] = sK[buf][l15 * 32 + 16 + a_dw(j, half)];
    st0 = wmma_bf16(a, bq1, st0);
#pragma unroll
    for (int j = 0; j < 8; ++j) a.u[j] = sK[buf][(16 + l15) * 32 + a_dw(j, half)];
    st1 = wmma_bf16(a, bq0, st1);
#pragma unroll
    for (int j = 0; j < 8; ++j) a.u[j] = sK[buf][(16 + l15) * 32 + 16 + a_dw(j, half)];
    st1 = wmma_bf16(a, bq1, st1);

    // online softmax for this lane's query (partner lane = lane^16)
    float mloc = -3.0e38f;
#pragma unroll
    for (int r = 0; r < 8; ++r) {
      st0[r] *= 0.125f;                  // 1/sqrt(HD)
      st1[r] *= 0.125f;
      mloc = fmaxf(mloc, fmaxf(st0[r], st1[r]));
    }
    mloc = fmaxf(mloc, __shfl_xor(mloc, 16));
    const float mnew = fmaxf(mrow, mloc);
    const float corr = __expf(mrow - mnew);
    float p0[8], p1[8], psum = 0.0f;
#pragma unroll
    for (int r = 0; r < 8; ++r) {
      p0[r] = __expf(st0[r] - mnew);
      p1[r] = __expf(st1[r] - mnew);
      psum += p0[r] + p1[r];
    }
    psum += __shfl_xor(psum, 16);
    lrow = lrow * corr + psum;
    mrow = mnew;
#pragma unroll
    for (int t = 0; t < 4; ++t)
#pragma unroll
      for (int r = 0; r < 8; ++r) oacc[t][r] *= corr;

    // build P^T B-fragment: kcol = e + 16*half; exchange halves with partner
    Frag16 pb;
#pragma unroll
    for (int r = 0; r < 8; ++r) {
      float send = half ? p0[r] : p1[r];
      float recv = __shfl_xor(send, 16);
      pb.h[r]     = f2bf(half ? recv  : p0[r]);
      pb.h[8 + r] = f2bf(half ? p1[r] : recv);
    }

    // O^T += V^T · P^T  (4 d-subtiles)
#pragma unroll
    for (int t = 0; t < 4; ++t) {
      Frag16 av;
#pragma unroll
      for (int j = 0; j < 8; ++j) av.u[j] = sV[buf][(t * 16 + l15) * 16 + a_dw(j, half)];
      oacc[t] = wmma_bf16(av, pb, oacc[t]);
    }
    __syncthreads();
  }

  // write o[b, q, h*64 + d] (8 consecutive bf16 per lane per subtile)
  const float inv = 1.0f / lrow;
  __bf16* orow = o + ((size_t)(bb * S + qrow)) * Hn + h * 64;
#pragma unroll
  for (int t = 0; t < 4; ++t) {
    unsigned int pk[4];
#pragma unroll
    for (int r = 0; r < 8; r += 2) {
      union { __bf16 h[2]; unsigned int u; } p;
      p.h[0] = f2bf(oacc[t][r] * inv);
      p.h[1] = f2bf(oacc[t][r + 1] * inv);
      pk[r >> 1] = p.u;
    }
    uint4 st; st.x = pk[0]; st.y = pk[1]; st.z = pk[2]; st.w = pk[3];
    *(uint4*)(orow + t * 16 + half * 8) = st;
  }
}

// ---------------------------------------------------------------------------
// Host launcher
// ---------------------------------------------------------------------------
extern "C" void kernel_launch(void* const* d_in, const int* in_sizes, int n_in,
                              void* d_out, int out_size, void* d_ws, size_t ws_size,
                              hipStream_t stream) {
  (void)in_sizes; (void)n_in; (void)out_size; (void)ws_size;
  const float* x    = (const float*)d_in[0];
  const float* ln1w = (const float*)d_in[1];
  const float* ln1b = (const float*)d_in[2];
  const float* Wqkv = (const float*)d_in[3];
  const float* bqkv = (const float*)d_in[4];
  const float* Wo   = (const float*)d_in[5];
  const float* bo   = (const float*)d_in[6];
  const float* ln2w = (const float*)d_in[7];
  const float* ln2b = (const float*)d_in[8];
  const float* fc1w = (const float*)d_in[9];
  const float* fc1b = (const float*)d_in[10];
  const float* fc2w = (const float*)d_in[11];
  const float* fc2b = (const float*)d_in[12];

  const size_t MiB = 1ull << 20;
  char* ws = (char*)d_ws;
  __bf16* h_bf  = (__bf16*)(ws + 0);          // 16 MiB   [8192 x 1024]
  __bf16* wqkvT = (__bf16*)(ws + 16 * MiB);   //  6 MiB   [3072 x 1024]
  __bf16* woT   = (__bf16*)(ws + 22 * MiB);   //  2 MiB   [1024 x 1024]
  __bf16* fc1T  = (__bf16*)(ws + 24 * MiB);   //  8 MiB   [4096 x 1024]
  __bf16* fc2T  = (__bf16*)(ws + 32 * MiB);   //  8 MiB   [1024 x 4096]
  __bf16* qkv   = (__bf16*)(ws + 40 * MiB);   // 48 MiB   [8192 x 3072]
  __bf16* ffn   = (__bf16*)(ws + 40 * MiB);   // 64 MiB   (reuses qkv region)
  __bf16* Vt    = (__bf16*)(ws + 104 * MiB);  // 16 MiB   [B,NH,HD,S]
  __bf16* o_bf  = (__bf16*)(ws + 120 * MiB);  // 16 MiB   [8192 x 1024]
  float*  x1    = (float*)(ws + 136 * MiB);   // 32 MiB   [8192 x 1024]

  // weights -> bf16, transposed to [N][K]
  cvtT_kernel<<<dim3(3072 / 256, 1024), 256, 0, stream>>>(Wqkv, wqkvT, 1024, 3072);
  cvtT_kernel<<<dim3(1024 / 256, 1024), 256, 0, stream>>>(Wo,   woT,   1024, 1024);
  cvtT_kernel<<<dim3(4096 / 256, 1024), 256, 0, stream>>>(fc1w, fc1T,  1024, 4096);
  cvtT_kernel<<<dim3(1024 / 256, 4096), 256, 0, stream>>>(fc2w, fc2T,  4096, 1024);

  // attention sub-block
  ln_bf16_kernel<<<8192, 256, 0, stream>>>(x, ln1w, ln1b, h_bf);
  gemm_wmma_kernel<OP_BIAS_BF16><<<dim3(64, 24), 256, 0, stream>>>(
      h_bf, wqkvT, bqkv, nullptr, qkv, 8192, 3072, 1024);
  vtrans_kernel<<<dim3(8, 64, 64), 256, 0, stream>>>(qkv, Vt);
  attn_kernel<<<dim3(32, 16, 4), 128, 0, stream>>>(qkv, Vt, o_bf);
  gemm_wmma_kernel<OP_RESID_F32><<<dim3(64, 8), 256, 0, stream>>>(
      o_bf, woT, bo, x, x1, 8192, 1024, 1024);

  // FFN sub-block
  ln_bf16_kernel<<<8192, 256, 0, stream>>>(x1, ln2w, ln2b, h_bf);
  gemm_wmma_kernel<OP_GELU_BF16><<<dim3(64, 32), 256, 0, stream>>>(
      h_bf, fc1T, fc1b, nullptr, ffn, 8192, 4096, 1024);
  gemm_wmma_kernel<OP_RESID_F32><<<dim3(64, 8), 256, 0, stream>>>(
      ffn, fc2T, fc2b, x1, d_out, 8192, 1024, 4096);
}
